// DenseBlock_36404142801573
// MI455X (gfx1250) — matrix operand order repaired and checked by reading, not compiled, not run on hardware
//
#include <hip/hip_runtime.h>
#include <hip/hip_bf16.h>
#include <math.h>

#define B_   16
#define C_   32
#define N_   500
#define T_   64
#define L_   4
#define E_   2
#define K_   3
#define NT_  (N_*T_)        /* 32000  */
#define BNT_ (B_*N_*T_)     /* 512000 */
#define S_   ((size_t)B_*C_*N_*T_)  /* 16,384,000 */
#define KP_  512            /* padded M/K for GEMM */

typedef __attribute__((ext_vector_type(16))) __bf16 v16bf;
typedef __attribute__((ext_vector_type(8)))  __bf16 v8bf;
typedef __attribute__((ext_vector_type(8)))  float  v8f;

__device__ __forceinline__ __bf16 f2bf(float f) {
  union { float f; unsigned u; } a; a.f = f;
  unsigned r = (a.u + 0x7FFFu + ((a.u >> 16) & 1u)) >> 16;   // RNE
  union { unsigned short s; __bf16 b; } o; o.s = (unsigned short)r;
  return o.b;
}
__device__ __forceinline__ float bf2f(__bf16 b) {
  union { unsigned short s; __bf16 b; } i; i.b = b;
  union { unsigned u; float f; } o; o.u = ((unsigned)i.s) << 16;
  return o.f;
}

// ---------------- BN statistics ----------------
__global__ void k_zero_stats(float* stats) {
  if (threadIdx.x < 64) stats[threadIdx.x] = 0.f;
}

__global__ void k_bn_reduce(const float* __restrict__ h, float* __restrict__ stats) {
  __shared__ float ss[256], ss2[256];
  int c = blockIdx.x >> 6;        // 64 blocks per channel
  int j = blockIdx.x & 63;
  float s = 0.f, s2 = 0.f;
  for (int idx = j * 256 + threadIdx.x; idx < BNT_; idx += 64 * 256) {
    int b = idx / NT_, rem = idx - b * NT_;
    float v = h[(size_t)(b * C_ + c) * NT_ + rem];
    s += v; s2 += v * v;
  }
  ss[threadIdx.x] = s; ss2[threadIdx.x] = s2;
  __syncthreads();
  for (int st = 128; st > 0; st >>= 1) {
    if (threadIdx.x < st) { ss[threadIdx.x] += ss[threadIdx.x + st]; ss2[threadIdx.x] += ss2[threadIdx.x + st]; }
    __syncthreads();
  }
  if (threadIdx.x == 0) { atomicAdd(&stats[c], ss[0]); atomicAdd(&stats[32 + c], ss2[0]); }
}

__global__ void k_bn_finalize(float* stats, const float* __restrict__ gamma,
                              const float* __restrict__ beta, int layer) {
  int c = threadIdx.x;
  if (c >= C_) return;
  float inv = 1.f / (float)BNT_;
  float m = stats[c] * inv;
  float v = stats[32 + c] * inv - m * m;
  float sc = gamma[layer * C_ + c] * rsqrtf(v + 1e-5f);
  stats[64 + c] = sc;
  stats[96 + c] = beta[layer * C_ + c] - m * sc;
}

// ------- fused BN/ReLU + causal convs + residual + gating -> gated x (bf16) -------
__global__ void k_gate(const float* __restrict__ h, const float* __restrict__ stats,
                       const float* __restrict__ rw, const float* __restrict__ rb,
                       const float* __restrict__ aw, const float* __restrict__ ab,
                       const float* __restrict__ gw, const float* __restrict__ gb,
                       __bf16* __restrict__ xg) {
  int pos = blockIdx.x * 256 + threadIdx.x;
  if (pos >= BNT_) return;
  int b = pos / NT_; int rem = pos - b * NT_;
  int n = rem / T_;  int t = rem - n * T_;
  size_t base = (size_t)b * C_ * NT_ + (size_t)n * T_ + t;
  float y0[C_], y1[C_], y2[C_], hr[C_];
  for (int c = 0; c < C_; c++) {
    float sc = stats[64 + c], sh = stats[96 + c];
    size_t off = base + (size_t)c * NT_;
    float hv = h[off];
    hr[c] = hv;
    float a0 = hv * sc + sh;          y0[c] = a0 > 0.f ? a0 : 0.f;
    if (t >= 1) { float a1 = h[off - 1] * sc + sh; y1[c] = a1 > 0.f ? a1 : 0.f; } else y1[c] = 0.f;
    if (t >= 2) { float a2 = h[off - 2] * sc + sh; y2[c] = a2 > 0.f ? a2 : 0.f; } else y2[c] = 0.f;
  }
  for (int o = 0; o < C_; o++) {
    float af = ab[o], ga = gb[o], re = rb[o];
    for (int c = 0; c < C_; c++) {
      re += hr[c] * rw[o * C_ + c];
      int wi = (o * C_ + c) * K_;     // k=0 -> t-2, k=1 -> t-1, k=2 -> t (causal)
      af += aw[wi] * y2[c] + aw[wi + 1] * y1[c] + aw[wi + 2] * y0[c];
      ga += gw[wi] * y2[c] + gw[wi + 1] * y1[c] + gw[wi + 2] * y0[c];
    }
    float v = tanhf(af + re) * (1.f / (1.f + expf(-ga)));
    xg[((size_t)(b * C_ + o) * N_ + n) * T_ + t] = f2bf(v);
  }
}

// ------- supports -> transposed, zero-padded bf16  ST[e][w][v] = A_e[v][w] -------
__global__ void k_prep_supports(const float* __restrict__ sup, __bf16* __restrict__ st) {
  int g = blockIdx.x * 256 + threadIdx.x;
  if (g >= E_ * KP_ * KP_) return;
  int e = g / (KP_ * KP_); int r = g - e * KP_ * KP_;
  int w = r >> 9; int v = r & (KP_ - 1);
  float val = (w < N_ && v < N_) ? sup[(size_t)e * N_ * N_ + (size_t)v * N_ + w] : 0.f;
  st[g] = f2bf(val);
}

// ------- graph diffusion: hs[b,e,c,w,t] = sum_v ST[e][w][v] * xg[b,c,v,t] -------
// WMMA bf16 GEMM: per (b,c,e,mblock): D[128x64] = A[128x512] * B[512x64]
__global__ void __launch_bounds__(256) k_graph_gemm(const __bf16* __restrict__ st,
                                                    const __bf16* __restrict__ xg,
                                                    float* __restrict__ hs) {
  __shared__ __align__(16) __bf16 As[128][32];   // [w-local][k-local]
  __shared__ __align__(16) __bf16 Bt[64][32];    // [t][k-local] (k contiguous)
  int bx = blockIdx.x;
  int mb = bx & 3; int e = (bx >> 2) & 1; int c = (bx >> 3) & 31; int b = bx >> 8;
  int tid  = threadIdx.x;
  int lane = tid & 31, wv = tid >> 5;            // 8 waves
  int lo = lane & 15, hi = lane >> 4;

  const __bf16* stp = st + (size_t)e * KP_ * KP_ + (size_t)mb * 128 * KP_;
  const __bf16* xgp = xg + (size_t)(b * C_ + c) * NT_;

  v8f acc[4];
  for (int i = 0; i < 4; i++)
    for (int j = 0; j < 8; j++) acc[i][j] = 0.f;

  int ar = tid >> 1, ah = (tid & 1) * 16;        // A staging: 16 bf16 per thread
  int bv = tid & 31, btc = (tid >> 5) * 8;       // B staging: 8 bf16 per thread

  for (int k0 = 0; k0 < KP_; k0 += 32) {
    __syncthreads();
    // stage A tile (128x32 bf16), ST is zero-padded so no masks needed
    const uint4* ga = (const uint4*)(stp + (size_t)ar * KP_ + k0 + ah);
    uint4 qa0 = ga[0], qa1 = ga[1];
    *(uint4*)&As[ar][ah]     = qa0;
    *(uint4*)&As[ar][ah + 8] = qa1;
    // stage B tile transposed into [t][k] so fragment reads are contiguous
    int v = k0 + bv;
    uint4 qb;
    if (v < N_) qb = *(const uint4*)(xgp + (size_t)v * T_ + btc);
    else { qb.x = 0u; qb.y = 0u; qb.z = 0u; qb.w = 0u; }
    const __bf16* qbp = (const __bf16*)&qb;
    #pragma unroll
    for (int j = 0; j < 8; j++) Bt[btc + j][bv] = qbp[j];
    __syncthreads();

    // A fragment (16x32 bf16): lanes hi=0 -> K 0..7 | 16..23 ; hi=1 -> K 8..15 | 24..31
    int row = wv * 16 + lo;
    v8bf a0 = *(const v8bf*)&As[row][hi * 8];
    v8bf a1 = *(const v8bf*)&As[row][16 + hi * 8];
    v16bf afrag = __builtin_shufflevector(a0, a1, 0,1,2,3,4,5,6,7,8,9,10,11,12,13,14,15);

    #pragma unroll
    for (int ns = 0; ns < 4; ns++) {
      // B fragment (32x16 bf16): column = t, hi=0 -> K 0..7 | 16..23 ; hi=1 -> K 8..15 | 24..31
      int ncol = ns * 16 + lo;
      v8bf b0 = *(const v8bf*)&Bt[ncol][hi * 8];
      v8bf b1 = *(const v8bf*)&Bt[ncol][16 + hi * 8];
      v16bf bfrag = __builtin_shufflevector(b0, b1, 0,1,2,3,4,5,6,7,8,9,10,11,12,13,14,15);
      acc[ns] = __builtin_amdgcn_wmma_f32_16x16x32_bf16(
          false, afrag, false, bfrag, (short)0, acc[ns], false, false);
    }
  }

  // D layout: VGPR j -> M = hi*8 + j, N = lo
  float* outp = hs + ((size_t)(b * E_ + e) * C_ + c) * NT_;
  #pragma unroll
  for (int ns = 0; ns < 4; ns++) {
    int tcol = ns * 16 + lo;
    #pragma unroll
    for (int j = 0; j < 8; j++) {
      int w = mb * 128 + wv * 16 + hi * 8 + j;
      if (w < N_) outp[(size_t)w * T_ + tcol] = acc[ns][j];
    }
  }
}

// ------- graph conv 96 -> 32 over concat [x, hs_e0, hs_e1] -------
__global__ void k_gconv(const __bf16* __restrict__ xg, const float* __restrict__ hs,
                        const float* __restrict__ gcw, const float* __restrict__ gcb,
                        float* __restrict__ hout) {
  int pos = blockIdx.x * 256 + threadIdx.x;
  if (pos >= BNT_) return;
  int b = pos / NT_; int rem = pos - b * NT_;
  float in[C_ * (E_ + 1)];
  for (int c = 0; c < C_; c++) in[c] = bf2f(xg[(size_t)(b * C_ + c) * NT_ + rem]);
  for (int e = 0; e < E_; e++)
    for (int c = 0; c < C_; c++)
      in[C_ + e * C_ + c] = hs[((size_t)(b * E_ + e) * C_ + c) * NT_ + rem];
  for (int o = 0; o < C_; o++) {
    float a = gcb[o];
    for (int cc = 0; cc < C_ * (E_ + 1); cc++) a += in[cc] * gcw[o * (C_ * (E_ + 1)) + cc];
    hout[(size_t)(b * C_ + o) * NT_ + rem] = a;
  }
}

// ------- fused self-attention over the L+1 stacked states -------
__global__ void k_attn(const float* __restrict__ x0, const float* __restrict__ stack,
                       const float* __restrict__ w1, const float* __restrict__ b1,
                       const float* __restrict__ w2, const float* __restrict__ b2,
                       const float* __restrict__ w3, const float* __restrict__ b3,
                       float* __restrict__ out) {
  int pos = blockIdx.x * 256 + threadIdx.x;
  if (pos >= BNT_) return;
  int b = pos / NT_; int rem = pos - b * NT_;
  size_t base = (size_t)b * C_ * NT_ + rem;
  float ev[L_ + 1];
  for (int l = 0; l <= L_; l++) {
    const float* hl = (l == 0) ? x0 : (stack + (size_t)(l - 1) * S_);
    float xs[C_];
    for (int c = 0; c < C_; c++) { float v = hl[base + (size_t)c * NT_]; xs[c] = v > 0.f ? v : 0.f; }
    float t1[64];
    for (int d = 0; d < 64; d++) {
      float a = b1[d];
      for (int c = 0; c < C_; c++) a += w1[d * C_ + c] * xs[c];
      t1[d] = a > 0.f ? a : 0.f;
    }
    float e3 = b3[0];
    for (int d2 = 0; d2 < 64; d2++) {
      float a = b2[d2];
      for (int d = 0; d < 64; d++) a += w2[d2 * 64 + d] * t1[d];
      e3 += w3[d2] * (a > 0.f ? a : 0.f);
    }
    ev[l] = e3;
  }
  float m = ev[0];
  for (int l = 1; l <= L_; l++) m = fmaxf(m, ev[l]);
  float wl[L_ + 1], sum = 0.f;
  for (int l = 0; l <= L_; l++) { wl[l] = expf(ev[l] - m); sum += wl[l]; }
  float inv = 1.f / sum;
  for (int c = 0; c < C_; c++) {
    float a = 0.f;
    for (int l = 0; l <= L_; l++) {
      const float* hl = (l == 0) ? x0 : (stack + (size_t)(l - 1) * S_);
      float v = hl[base + (size_t)c * NT_];
      a += wl[l] * (v > 0.f ? v : 0.f);
    }
    out[base + (size_t)c * NT_] = a * inv;
  }
}

extern "C" void kernel_launch(void* const* d_in, const int* in_sizes, int n_in,
                              void* d_out, int out_size, void* d_ws, size_t ws_size,
                              hipStream_t stream) {
  (void)in_sizes; (void)n_in; (void)out_size; (void)ws_size;
  const float* x      = (const float*)d_in[0];
  const float* sup    = (const float*)d_in[1];
  const float* gamma  = (const float*)d_in[2];
  const float* beta   = (const float*)d_in[3];
  const float* res_w  = (const float*)d_in[4];
  const float* res_b  = (const float*)d_in[5];
  const float* aff_w  = (const float*)d_in[6];
  const float* aff_b  = (const float*)d_in[7];
  const float* gate_w = (const float*)d_in[8];
  const float* gate_b = (const float*)d_in[9];
  const float* gc_w   = (const float*)d_in[10];
  const float* gc_b   = (const float*)d_in[11];
  const float* w1 = (const float*)d_in[12];
  const float* b1 = (const float*)d_in[13];
  const float* w2 = (const float*)d_in[14];
  const float* b2 = (const float*)d_in[15];
  const float* w3 = (const float*)d_in[16];
  const float* b3 = (const float*)d_in[17];

  // workspace layout (all offsets 256B aligned)
  char* ws = (char*)d_ws;
  float*  stack = (float*)ws;                          // 4*S f32 = 262,144,000 B
  __bf16* xg    = (__bf16*)(ws + 262144000ull);        // S bf16  =  32,768,000 B
  float*  hs    = (float*)(ws + 294912000ull);         // 2*S f32 = 131,072,000 B
  __bf16* stT   = (__bf16*)(ws + 425984000ull);        // 2*512*512 bf16 = 1 MiB
  float*  stats = (float*)(ws + 427032576ull);         // 128 f32

  k_prep_supports<<<(E_ * KP_ * KP_ + 255) / 256, 256, 0, stream>>>(sup, stT);

  const int posBlocks = (BNT_ + 255) / 256;            // 2000
  for (int i = 0; i < L_; i++) {
    const float* hin = (i == 0) ? x : (stack + (size_t)(i - 1) * S_);
    float* hout = stack + (size_t)i * S_;
    k_zero_stats<<<1, 64, 0, stream>>>(stats);
    k_bn_reduce<<<C_ * 64, 256, 0, stream>>>(hin, stats);
    k_bn_finalize<<<1, 32, 0, stream>>>(stats, gamma, beta, i);
    k_gate<<<posBlocks, 256, 0, stream>>>(hin, stats,
        res_w + i * C_ * C_,       res_b + i * C_,
        aff_w + i * C_ * C_ * K_,  aff_b + i * C_,
        gate_w + i * C_ * C_ * K_, gate_b + i * C_, xg);
    k_graph_gemm<<<4 * E_ * C_ * B_, 256, 0, stream>>>(stT, xg, hs);
    k_gconv<<<posBlocks, 256, 0, stream>>>(xg, hs,
        gc_w + i * C_ * C_ * (E_ + 1), gc_b + i * C_, hout);
  }
  k_attn<<<posBlocks, 256, 0, stream>>>(x, stack, w1, b1, w2, b2, w3, b3, (float*)d_out);
}
